// HMABottleneck_43190191129253
// MI455X (gfx1250) — matrix operand order repaired and checked by reading, not compiled, not run on hardware
//
#include <hip/hip_runtime.h>

// ---------------------------------------------------------------------------
// MI455X (gfx1250) implementation.  All matmul-shaped work (projections,
// x_dbl/delta einsums, im2col'd 3x3/1x1 convs) runs through
// v_wmma_f32_16x16x32_bf16 (f32 -> bf16 staged in LDS, f32 accum).
// ---------------------------------------------------------------------------

#define BNS 0.99999500003749969f   // 1/sqrt(1+1e-5)

typedef __attribute__((ext_vector_type(16))) __bf16 v16bf;
typedef __attribute__((ext_vector_type(8)))  __bf16 v8bf;
typedef __attribute__((ext_vector_type(4)))  __bf16 v4bf;
typedef __attribute__((ext_vector_type(8)))  float  v8f;

__device__ __forceinline__ float gelu_f(float x) {
    return 0.5f * x * (1.0f + erff(x * 0.7071067811865475f));
}
__device__ __forceinline__ float silu_f(float x) {
    return x / (1.0f + __expf(-x));
}
__device__ __forceinline__ float softplus_f(float x) {
    return (x > 20.f) ? x : log1pf(__expf(x));
}

// ---------------------------------------------------------------------------
// Generic WMMA GEMM:  D[M,N] = epi( A[M,K] * W[N,K]^T )
// 128 threads = 4 wave32, wave grid 2x2, block tile 64x64, K-tile 32.
// LDS tiles are bf16 [64][40] (stride 40 -> 80B rows: 16B-aligned b128
// fragment loads, conflict-free bank spread).  Ping-pong LDS + register
// prefetch -> one barrier per K-step, branchless staging (rows clamped,
// out-of-range results masked at the epilogue store).
// epi: 0 none(row-major) 1 bias+softplus(row-major)
//      2 NCHW gelu(bn)   3 NCHW gelu(bn+res)  4 NCHW gelu(bn)+res  5 NCHW +res
// ---------------------------------------------------------------------------
#define LDSS 40

__device__ __forceinline__ void stage_load(const float* __restrict__ base,
                                           int rowBase, int rowMax, int ld,
                                           int k0, int tid, float4* reg)
{
#pragma unroll
    for (int i = 0; i < 4; ++i) {
        int idx = tid + i * 128;
        int r = idx >> 3, c4 = (idx & 7) * 4;
        int gr = rowBase + r;
        gr = (gr < rowMax) ? gr : (rowMax - 1);          // branchless clamp
        reg[i] = *(const float4*)(base + (long)gr * ld + k0 + c4);
    }
}

__device__ __forceinline__ void stage_store(const float4* reg, __bf16* lds, int tid)
{
#pragma unroll
    for (int i = 0; i < 4; ++i) {
        int idx = tid + i * 128;
        int r = idx >> 3, c4 = (idx & 7) * 4;
        v4bf o;
        o[0] = (__bf16)reg[i].x; o[1] = (__bf16)reg[i].y;
        o[2] = (__bf16)reg[i].z; o[3] = (__bf16)reg[i].w;
        *(v4bf*)(&lds[r * LDSS + c4]) = o;               // 8B ds_store
    }
}

__global__ __launch_bounds__(128)
void wmma_gemm_kernel(const float* __restrict__ A, const float* __restrict__ Wm,
                      float* __restrict__ Co,
                      int M, int N, int K, int lda, int ldw, int ldc,
                      long sA, long sW, long sC, int wmod,
                      const float* __restrict__ bias, long sBias,
                      const float* __restrict__ scale, const float* __restrict__ shift,
                      const float* __restrict__ res, int HW, int epi)
{
    __shared__ __bf16 ldsA[2][64 * LDSS];
    __shared__ __bf16 ldsB[2][64 * LDSS];

    const int z  = blockIdx.z;
    const int zw = wmod ? (z % wmod) : z;
    A  += (long)z  * sA;
    Wm += (long)zw * sW;
    Co += (long)z  * sC;
    const float* biasz = bias ? (bias + (long)zw * sBias) : nullptr;

    const int mBase = blockIdx.y * 64;
    const int nBase = blockIdx.x * 64;
    const int tid   = threadIdx.x;
    const int wv    = tid >> 5;
    const int lane  = tid & 31;
    const int wm    = wv >> 1;
    const int wn    = wv & 1;
    const int half  = lane >> 4;
    const int mn    = lane & 15;

    v8f acc[2][2];
#pragma unroll
    for (int fm = 0; fm < 2; ++fm)
#pragma unroll
        for (int fn = 0; fn < 2; ++fn)
#pragma unroll
            for (int e = 0; e < 8; ++e) acc[fm][fn][e] = 0.f;

    float4 ra[4], rw[4];
    stage_load(A,  mBase, M, lda, 0, tid, ra);
    stage_load(Wm, nBase, N, ldw, 0, tid, rw);
    stage_store(ra, &ldsA[0][0], tid);
    stage_store(rw, &ldsB[0][0], tid);
    __syncthreads();

    const int nk = K >> 5;
    for (int it = 0; it < nk; ++it) {
        const int buf = it & 1;
        const bool more = (it + 1 < nk);
        if (more) {                                   // uniform branch
            stage_load(A,  mBase, M, lda, (it + 1) << 5, tid, ra);
            stage_load(Wm, nBase, N, ldw, (it + 1) << 5, tid, rw);
        }

        v16bf af[2], bfv[2];
#pragma unroll
        for (int f = 0; f < 2; ++f) {
            // A fragment: lane holds row m=lane&15; K = 16*(e>>3)+8*half+(e&7)
            const __bf16* p = &ldsA[buf][(wm * 32 + f * 16 + mn) * LDSS];
            v8bf alo = *(const v8bf*)(p + 8 * half);
            v8bf ahi = *(const v8bf*)(p + 16 + 8 * half);
            af[f] = __builtin_shufflevector(alo, ahi, 0,1,2,3,4,5,6,7,8,9,10,11,12,13,14,15);
            // B fragment: lane holds col n=lane&15; K = 16*(e>>3)+8*half+(e&7)
            const __bf16* q = &ldsB[buf][(wn * 32 + f * 16 + mn) * LDSS];
            v8bf blo = *(const v8bf*)(q + 8 * half);
            v8bf bhi = *(const v8bf*)(q + 16 + 8 * half);
            bfv[f] = __builtin_shufflevector(blo, bhi, 0,1,2,3,4,5,6,7,8,9,10,11,12,13,14,15);
        }
#pragma unroll
        for (int fm = 0; fm < 2; ++fm)
#pragma unroll
            for (int fn = 0; fn < 2; ++fn)
                acc[fm][fn] = __builtin_amdgcn_wmma_f32_16x16x32_bf16(
                    false, af[fm], false, bfv[fn], (short)0, acc[fm][fn], false, false);

        if (more) {
            stage_store(ra, &ldsA[buf ^ 1][0], tid);
            stage_store(rw, &ldsB[buf ^ 1][0], tid);
        }
        __syncthreads();
    }

    // epilogue (C/D layout: VGPR v, lane l -> M = 8*(l>>4)+v, N = l&15)
#pragma unroll
    for (int fm = 0; fm < 2; ++fm)
#pragma unroll
        for (int fn = 0; fn < 2; ++fn)
#pragma unroll
            for (int v = 0; v < 8; ++v) {
                int gm = mBase + wm * 32 + fm * 16 + half * 8 + v;
                int gn = nBase + wn * 32 + fn * 16 + mn;
                if (gm >= M || gn >= N) continue;
                float val = acc[fm][fn][v];
                if (epi == 0) {
                    Co[(long)gm * ldc + gn] = val;
                } else if (epi == 1) {
                    Co[(long)gm * ldc + gn] = softplus_f(val + biasz[gn]);
                } else {
                    int bimg = gm / HW, p = gm % HW;
                    long oidx = ((long)bimg * N + gn) * HW + p;
                    if (epi == 2)       val = gelu_f(val * scale[gn] * BNS + shift[gn]);
                    else if (epi == 3)  val = gelu_f(val * scale[gn] * BNS + shift[gn] + res[oidx]);
                    else if (epi == 4)  val = gelu_f(val * scale[gn] * BNS + shift[gn]) + res[oidx];
                    else                val = val + res[oidx];           // epi 5
                    Co[oidx] = val;
                }
            }
}

// ---------------------------------------------------------------------------
// LayerNorm over C, NCHW input -> NHWC output.  One block per (b, pixel).
// ---------------------------------------------------------------------------
__global__ __launch_bounds__(256)
void ln_nchw_kernel(const float* __restrict__ x, const float* __restrict__ w,
                    const float* __restrict__ b, float* __restrict__ out,
                    int C, int HW, float eps)
{
    __shared__ float s1[256], s2[256];
    int p = blockIdx.x, bimg = p / HW, pix = p % HW;
    const float* xb = x + (long)bimg * C * HW + pix;
    float sum = 0.f, sq = 0.f;
    for (int c = threadIdx.x; c < C; c += 256) {
        float v = xb[(long)c * HW]; sum += v; sq += v * v;
    }
    s1[threadIdx.x] = sum; s2[threadIdx.x] = sq; __syncthreads();
    for (int s = 128; s > 0; s >>= 1) {
        if (threadIdx.x < s) { s1[threadIdx.x] += s1[threadIdx.x + s]; s2[threadIdx.x] += s2[threadIdx.x + s]; }
        __syncthreads();
    }
    float mu = s1[0] / C;
    float rs = rsqrtf(s2[0] / C - mu * mu + eps);
    for (int c = threadIdx.x; c < C; c += 256) {
        float v = xb[(long)c * HW];
        out[(long)p * C + c] = (v - mu) * rs * w[c] + b[c];
    }
}

// ---------------------------------------------------------------------------
// Depthwise 3x3 (pad1) on NHWC xz[:, :DI] + bias + SiLU, scattered into the
// 4 scan-direction layouts xsT (B,4,L,DI) (d contiguous for coalescing).
// ---------------------------------------------------------------------------
__global__ void dwconv_silu_scatter_kernel(const float* __restrict__ xz,
                                           const float* __restrict__ wgt,
                                           const float* __restrict__ bias,
                                           float* __restrict__ xsT,
                                           int H, int W, int DI, long total)
{
    long i = (long)blockIdx.x * blockDim.x + threadIdx.x;
    if (i >= total) return;
    int d = i % DI; long t = i / DI;
    int w = t % W; t /= W;
    int h = t % H; int b = (int)(t / H);
    int C2 = 2 * DI;
    float acc = bias[d];
#pragma unroll
    for (int kh = 0; kh < 3; ++kh)
#pragma unroll
        for (int kw = 0; kw < 3; ++kw) {
            int hh = h + kh - 1, ww = w + kw - 1;
            if (hh >= 0 && hh < H && ww >= 0 && ww < W)
                acc += wgt[d * 9 + kh * 3 + kw] *
                       xz[((long)(b * H + hh) * W + ww) * C2 + d];
        }
    float s = silu_f(acc);
    int L = H * W, l = h * W + w, lT = w * H + h;
    long base = (long)b * 4 * L * DI;
    xsT[base + (long)(0 * L + l)            * DI + d] = s;
    xsT[base + (long)(1 * L + lT)           * DI + d] = s;
    xsT[base + (long)(2 * L + (L - 1 - l))  * DI + d] = s;
    xsT[base + (long)(3 * L + (L - 1 - lT)) * DI + d] = s;
}

// ---------------------------------------------------------------------------
// Selective scan: one thread per (b,k,d); N=16 state in registers.
// u/delta read from (B,4,L,DI) layouts; B_l/C_l broadcast from x_dblT cols 32..63.
// ---------------------------------------------------------------------------
__global__ __launch_bounds__(256)
void scan_kernel(const float* __restrict__ xsT, const float* __restrict__ dT,
                 const float* __restrict__ xdbl, const float* __restrict__ Alog,
                 const float* __restrict__ Dp, float* __restrict__ ysT,
                 int K, int L, int DI)
{
    int bk = blockIdx.x;
    int k  = bk % K;
    int d  = blockIdx.y * blockDim.x + threadIdx.x;
    float a[16], h[16];
#pragma unroll
    for (int n = 0; n < 16; ++n) {
        a[n] = -__expf(Alog[((long)k * DI + d) * 16 + n]);
        h[n] = 0.f;
    }
    float Dv = Dp[(long)k * DI + d];
    long ubase = (long)bk * L * DI;
    long cbase = (long)bk * L * 64;
    for (int l = 0; l < L; ++l) {
        float u  = xsT[ubase + (long)l * DI + d];
        float dl = dT [ubase + (long)l * DI + d];
        const float4* bp = (const float4*)(xdbl + cbase + (long)l * 64 + 32);
        float4 B0 = bp[0], B1 = bp[1], B2 = bp[2], B3 = bp[3];
        float4 C0 = bp[4], C1 = bp[5], C2v = bp[6], C3 = bp[7];
        float Bv[16] = {B0.x,B0.y,B0.z,B0.w, B1.x,B1.y,B1.z,B1.w,
                        B2.x,B2.y,B2.z,B2.w, B3.x,B3.y,B3.z,B3.w};
        float Cv[16] = {C0.x,C0.y,C0.z,C0.w, C1.x,C1.y,C1.z,C1.w,
                        C2v.x,C2v.y,C2v.z,C2v.w, C3.x,C3.y,C3.z,C3.w};
        float du = dl * u, y = 0.f;
#pragma unroll
        for (int n = 0; n < 16; ++n) {
            h[n] = __expf(dl * a[n]) * h[n] + du * Bv[n];
            y += h[n] * Cv[n];
        }
        ysT[ubase + (long)l * DI + d] = y + Dv * u;
    }
}

// ---------------------------------------------------------------------------
// 4-direction combine + LayerNorm(DI) + silu(z) gate -> yg (B*L, DI).
// One block per (b,l); DI=1024 -> 4 elems/thread.
// ---------------------------------------------------------------------------
__global__ __launch_bounds__(256)
void combine_ln_gate_kernel(const float* __restrict__ ysT, const float* __restrict__ xz,
                            const float* __restrict__ onw, const float* __restrict__ onb,
                            float* __restrict__ yg, int H, int W, int DI)
{
    __shared__ float s1[256], s2[256];
    int L = H * W, p = blockIdx.x, b = p / L, l = p % L;
    int hh = l / W, ww = l % W, lT = ww * H + hh;
    long base = (long)b * 4 * L * DI;
    float vals[4]; float sum = 0.f, sq = 0.f;
#pragma unroll
    for (int i = 0; i < 4; ++i) {
        int d = threadIdx.x + i * 256;
        float v = ysT[base + (long)(0 * L + l)            * DI + d]
                + ysT[base + (long)(1 * L + lT)           * DI + d]
                + ysT[base + (long)(2 * L + (L - 1 - l))  * DI + d]
                + ysT[base + (long)(3 * L + (L - 1 - lT)) * DI + d];
        vals[i] = v; sum += v; sq += v * v;
    }
    s1[threadIdx.x] = sum; s2[threadIdx.x] = sq; __syncthreads();
    for (int s = 128; s > 0; s >>= 1) {
        if (threadIdx.x < s) { s1[threadIdx.x] += s1[threadIdx.x + s]; s2[threadIdx.x] += s2[threadIdx.x + s]; }
        __syncthreads();
    }
    float mu = s1[0] / DI;
    float rs = rsqrtf(s2[0] / DI - mu * mu + 1e-5f);
#pragma unroll
    for (int i = 0; i < 4; ++i) {
        int d = threadIdx.x + i * 256;
        float zv = xz[(long)p * 2 * DI + DI + d];
        yg[(long)p * DI + d] = ((vals[i] - mu) * rs * onw[d] + onb[d]) * silu_f(zv);
    }
}

// ---------------------------------------------------------------------------
// im2col (also handles 1x1/pad0 for pointwise convs): out[M, Cin*KH*KW]
// ---------------------------------------------------------------------------
__global__ void im2col_kernel(const float* __restrict__ in, float* __restrict__ out,
                              int B, int Cin, int H, int W, int KH, int KW, int pad)
{
    long total = (long)B * H * W * Cin * KH * KW;
    long i = (long)blockIdx.x * blockDim.x + threadIdx.x;
    if (i >= total) return;
    int Kc = Cin * KH * KW;
    int kk = (int)(i % Kc); long m = i / Kc;
    int w = (int)(m % W); long t = m / W;
    int h = (int)(t % H); int b = (int)(t / H);
    int kw = kk % KW, kh = (kk / KW) % KH, ci = kk / (KW * KH);
    int hh = h + kh - pad, ww = w + kw - pad;
    float v = 0.f;
    if (hh >= 0 && hh < H && ww >= 0 && ww < W)
        v = in[((long)(b * Cin + ci) * H + hh) * W + ww];
    out[m * Kc + kk] = v;
}

// depthwise 3x3 stride-2 pad-1 + BN + GELU (downsample stage 1), NCHW
__global__ void dw_s2_bn_gelu_kernel(const float* __restrict__ in, const float* __restrict__ wgt,
                                     const float* __restrict__ sc, const float* __restrict__ sh,
                                     float* __restrict__ out, int C, int H, int W, long total)
{
    long i = (long)blockIdx.x * blockDim.x + threadIdx.x;
    if (i >= total) return;
    int Ho = H / 2, Wo = W / 2;
    int wo = (int)(i % Wo); long t = i / Wo;
    int ho = (int)(t % Ho); t /= Ho;
    int c = (int)(t % C); int b = (int)(t / C);
    float acc = 0.f;
#pragma unroll
    for (int kh = 0; kh < 3; ++kh)
#pragma unroll
        for (int kw = 0; kw < 3; ++kw) {
            int hh = 2 * ho + kh - 1, ww = 2 * wo + kw - 1;
            if (hh >= 0 && hh < H && ww >= 0 && ww < W)
                acc += wgt[c * 9 + kh * 3 + kw] * in[((long)(b * C + c) * H + hh) * W + ww];
        }
    out[((long)(b * C + c) * Ho + ho) * Wo + wo] = gelu_f(acc * sc[c] * BNS + sh[c]);
}

__global__ void bn_gelu_kernel(const float* __restrict__ in, const float* __restrict__ sc,
                               const float* __restrict__ sh, float* __restrict__ out,
                               int C, int HW, long total)
{
    long i = (long)blockIdx.x * blockDim.x + threadIdx.x;
    if (i >= total) return;
    int c = (int)((i / HW) % C);
    out[i] = gelu_f(in[i] * sc[c] * BNS + sh[c]);
}

// bilinear upsample (linspace 0..h-1 over Ho points) + add skip, NCHW
__global__ void upsample_add_kernel(const float* __restrict__ in, const float* __restrict__ add,
                                    float* __restrict__ out, int C, int h, int w,
                                    int Ho, int Wo, long total)
{
    long i = (long)blockIdx.x * blockDim.x + threadIdx.x;
    if (i >= total) return;
    int wo = (int)(i % Wo); long t = i / Wo;
    int ho = (int)(t % Ho); t /= Ho;
    int c = (int)(t % C); int b = (int)(t / C);
    float fy = (Ho > 1) ? (float)ho * (float)(h - 1) / (float)(Ho - 1) : 0.f;
    float fx = (Wo > 1) ? (float)wo * (float)(w - 1) / (float)(Wo - 1) : 0.f;
    int y0 = (int)floorf(fy); int y1 = (y0 + 1 < h) ? y0 + 1 : h - 1; float wy = fy - (float)y0;
    int x0 = (int)floorf(fx); int x1 = (x0 + 1 < w) ? x0 + 1 : w - 1; float wx = fx - (float)x0;
    const float* basep = in + (long)(b * C + c) * h * w;
    float g0 = basep[y0 * w + x0] * (1.f - wx) + basep[y0 * w + x1] * wx;
    float g1 = basep[y1 * w + x0] * (1.f - wx) + basep[y1 * w + x1] * wx;
    out[i] = g0 * (1.f - wy) + g1 * wy + add[i];
}

// ---------------------------------------------------------------------------
// Host orchestration
// ---------------------------------------------------------------------------
extern "C" void kernel_launch(void* const* d_in, const int* in_sizes, int n_in,
                              void* d_out, int out_size, void* d_ws, size_t ws_size,
                              hipStream_t stream)
{
    (void)in_sizes; (void)n_in; (void)out_size;
    const int B = 4, C = 512, DI = 1024, KD = 4, R = 32;

    const float* x        = (const float*)d_in[0];
    const float* ln_w     = (const float*)d_in[1];
    const float* ln_b     = (const float*)d_in[2];
    const float* in_proj  = (const float*)d_in[3];
    const float* conv_w   = (const float*)d_in[4];
    const float* conv_b   = (const float*)d_in[5];
    const float* xproj    = (const float*)d_in[6];
    const float* dtw      = (const float*)d_in[7];
    const float* dtb      = (const float*)d_in[8];
    const float* Alog     = (const float*)d_in[9];
    const float* Dp       = (const float*)d_in[10];
    const float* onw      = (const float*)d_in[11];
    const float* onb      = (const float*)d_in[12];
    const float* outproj  = (const float*)d_in[13];
    const float* ds_dw    = (const float*)d_in[14];
    const float* ds_bn1s  = (const float*)d_in[15];
    const float* ds_bn1b  = (const float*)d_in[16];
    const float* ds_pw    = (const float*)d_in[17];
    const float* ds_bn2s  = (const float*)d_in[18];
    const float* ds_bn2b  = (const float*)d_in[19];
    const float* fus_c1   = (const float*)d_in[20];
    const float* fus_bn1s = (const float*)d_in[21];
    const float* fus_bn1b = (const float*)d_in[22];
    const float* fus_c2   = (const float*)d_in[23];
    const float* fus_bn2s = (const float*)d_in[24];
    const float* fus_bn2b = (const float*)d_in[25];
    const float* op_bn0s  = (const float*)d_in[26];
    const float* op_bn0b  = (const float*)d_in[27];
    const float* op_conv  = (const float*)d_in[28];
    const float* op_bn1s  = (const float*)d_in[29];
    const float* op_bn1b  = (const float*)d_in[30];
    float* out = (float*)d_out;

    // bump allocator over d_ws
    uintptr_t wbase = (uintptr_t)d_ws;
    size_t off = 0;
    auto alloc = [&](size_t nfloats) -> float* {
        float* p = (float*)(wbase + off);
        off += ((nfloats * sizeof(float)) + 255) & ~(size_t)255;
        return p;
    };

    // persistent buffers
    float* v0      = alloc(524288);
    float* xr      = alloc(524288);
    float* p1      = alloc(131072);
    float* p2      = alloc(32768);
    float* proc0   = alloc(524288);
    float* proc1   = alloc(131072);
    float* proc2   = alloc(32768);
    float* t_h8    = alloc(131072);
    float* t_h4    = alloc(32768);
    float* t_h16   = alloc(524288);
    float* up8     = alloc(131072);
    float* fused8  = alloc(131072);
    float* up16    = alloc(524288);
    float* fused16 = alloc(524288);
    float* hbuf    = alloc(524288);
    // per-VSS temporaries (max = full resolution)
    float* t_ln   = alloc(524288);
    float* t_xz   = alloc(2097152);
    float* t_xsT  = alloc(4194304);
    float* t_xdbl = alloc(262144);
    float* t_delta= alloc(4194304);
    float* t_ysT  = alloc(4194304);
    float* t_yg   = alloc(1048576);
    float* t_col  = t_ln;   // im2col buffer aliased over VSS temps (never live together)
    if (off > ws_size) return;

    auto gemm = [&](const float* Ap, const float* Wp, float* Cp,
                    int M, int Nn, int Kk, int lda, int ldw, int ldc,
                    long sAb, long sWb, long sCb, int batch, int wmod,
                    const float* bias, long sBias,
                    const float* sc, const float* sh,
                    const float* res, int HW, int epi) {
        dim3 g((unsigned)((Nn + 63) / 64), (unsigned)((M + 63) / 64), (unsigned)batch);
        wmma_gemm_kernel<<<g, 128, 0, stream>>>(Ap, Wp, Cp, M, Nn, Kk, lda, ldw, ldc,
                                                sAb, sWb, sCb, wmod, bias, sBias,
                                                sc, sh, res, HW, epi);
    };

    auto vss = [&](const float* xc, float* outb, int i, int H, int W) {
        const int L = H * W, M = B * L;
        ln_nchw_kernel<<<M, 256, 0, stream>>>(xc, ln_w + (long)i * C, ln_b + (long)i * C,
                                              t_ln, C, L, 1e-6f);
        gemm(t_ln, in_proj + (long)i * 2 * DI * C, t_xz, M, 2 * DI, C, C, C, 2 * DI,
             0, 0, 0, 1, 0, nullptr, 0, nullptr, nullptr, nullptr, 0, 0);
        {
            long tot = (long)M * DI;
            dwconv_silu_scatter_kernel<<<(unsigned)((tot + 255) / 256), 256, 0, stream>>>(
                t_xz, conv_w + (long)i * DI * 9, conv_b + (long)i * DI, t_xsT, H, W, DI, tot);
        }
        gemm(t_xsT, xproj + (long)i * KD * 64 * DI, t_xdbl, L, 64, DI, DI, DI, 64,
             (long)L * DI, (long)64 * DI, (long)L * 64, B * KD, KD,
             nullptr, 0, nullptr, nullptr, nullptr, 0, 0);
        gemm(t_xdbl, dtw + (long)i * KD * DI * R, t_delta, L, DI, R, 64, R, DI,
             (long)L * 64, (long)DI * R, (long)L * DI, B * KD, KD,
             dtb + (long)i * KD * DI, DI, nullptr, nullptr, nullptr, 0, 1);
        scan_kernel<<<dim3((unsigned)(B * KD), (unsigned)(DI / 256)), 256, 0, stream>>>(
            t_xsT, t_delta, t_xdbl, Alog + (long)i * KD * DI * 16, Dp + (long)i * KD * DI,
            t_ysT, KD, L, DI);
        combine_ln_gate_kernel<<<M, 256, 0, stream>>>(t_ysT, t_xz, onw + (long)i * DI,
                                                      onb + (long)i * DI, t_yg, H, W, DI);
        gemm(t_yg, outproj + (long)i * C * DI, outb, M, C, DI, DI, DI, C,
             0, 0, 0, 1, 0, nullptr, 0, nullptr, nullptr, xc, L, 5);
    };

    auto down = [&](const float* xc, float* outb, int j, int H, int W, float* hb) {
        int Ho = H / 2, Wo = W / 2;
        long tot = (long)B * C * Ho * Wo;
        dw_s2_bn_gelu_kernel<<<(unsigned)((tot + 255) / 256), 256, 0, stream>>>(
            xc, ds_dw + (long)j * C * 9, ds_bn1s + (long)j * C, ds_bn1b + (long)j * C,
            hb, C, H, W, tot);
        long tc = (long)B * Ho * Wo * C;
        im2col_kernel<<<(unsigned)((tc + 255) / 256), 256, 0, stream>>>(hb, t_col, B, C, Ho, Wo, 1, 1, 0);
        gemm(t_col, ds_pw + (long)j * C * C, outb, B * Ho * Wo, C, C, C, C, C,
             0, 0, 0, 1, 0, nullptr, 0, ds_bn2s + (long)j * C, ds_bn2b + (long)j * C,
             nullptr, Ho * Wo, 2);
    };

    auto resblk = [&](const float* xc, float* outb, int j, int H, int W, float* hb) {
        long tc = (long)B * H * W * C * 9;
        im2col_kernel<<<(unsigned)((tc + 255) / 256), 256, 0, stream>>>(xc, t_col, B, C, H, W, 3, 3, 1);
        gemm(t_col, fus_c1 + (long)j * C * C * 9, hb, B * H * W, C, C * 9, C * 9, C * 9, C,
             0, 0, 0, 1, 0, nullptr, 0, fus_bn1s + (long)j * C, fus_bn1b + (long)j * C,
             nullptr, H * W, 2);
        im2col_kernel<<<(unsigned)((tc + 255) / 256), 256, 0, stream>>>(hb, t_col, B, C, H, W, 3, 3, 1);
        gemm(t_col, fus_c2 + (long)j * C * C * 9, outb, B * H * W, C, C * 9, C * 9, C * 9, C,
             0, 0, 0, 1, 0, nullptr, 0, fus_bn2s + (long)j * C, fus_bn2b + (long)j * C,
             xc, H * W, 3);
    };

    // ---- forward ----
    vss(x,  v0, 0, 16, 16);
    vss(v0, xr, 1, 16, 16);
    down(xr, p1, 0, 16, 16, t_h8);
    down(p1, p2, 1, 8, 8, t_h4);
    vss(xr, proc0, 2, 16, 16);
    vss(p1, proc1, 3, 8, 8);
    vss(p2, proc2, 4, 4, 4);

    {
        long tot = (long)B * C * 8 * 8;
        upsample_add_kernel<<<(unsigned)((tot + 255) / 256), 256, 0, stream>>>(
            proc2, proc1, up8, C, 4, 4, 8, 8, tot);
    }
    resblk(up8, fused8, 0, 8, 8, t_h8);
    {
        long tot = (long)B * C * 16 * 16;
        upsample_add_kernel<<<(unsigned)((tot + 255) / 256), 256, 0, stream>>>(
            fused8, proc0, up16, C, 8, 8, 16, 16, tot);
    }
    resblk(up16, fused16, 1, 16, 16, t_h16);

    {
        long tot = (long)B * C * 256;
        bn_gelu_kernel<<<(unsigned)((tot + 255) / 256), 256, 0, stream>>>(
            fused16, op_bn0s, op_bn0b, hbuf, C, 256, tot);
        long tc = (long)B * 256 * C * 9;
        im2col_kernel<<<(unsigned)((tc + 255) / 256), 256, 0, stream>>>(hbuf, t_col, B, C, 16, 16, 3, 3, 1);
        gemm(t_col, op_conv, out, B * 256, C, C * 9, C * 9, C * 9, C,
             0, 0, 0, 1, 0, nullptr, 0, op_bn1s, op_bn1b, x, 256, 4);
    }
}
